// INENHead_86698209837427
// MI455X (gfx1250) — matrix operand order repaired and checked by reading, not compiled, not run on hardware
//
#include <hip/hip_runtime.h>

// GCN: 3x [20000x256 @ 256x256] GEMM via v_wmma_f32_16x16x32_bf16 (fp32 accum),
// segment-sum via global_atomic_add_f32, small 256->3 classifier on VALU.

#define NNODES 20000
#define NEDGES 320000
#define D 256

typedef __attribute__((ext_vector_type(16))) __bf16 v16bf;
typedef __attribute__((ext_vector_type(8)))  float  v8f;

// ---------------- degree / normalization ----------------
__global__ void deg_init_k(float* __restrict__ deg, int n) {
    int i = blockIdx.x * blockDim.x + threadIdx.x;
    if (i < n) deg[i] = 1.0f;                    // self-loop contributes 1
}

__global__ void deg_edges_k(const int* __restrict__ dst, float* __restrict__ deg, int e) {
    int i = blockIdx.x * blockDim.x + threadIdx.x;
    if (i < e) atomicAdd(&deg[dst[i]], 1.0f);
}

__global__ void rsqrt_k(float* __restrict__ deg, int n) {
    int i = blockIdx.x * blockDim.x + threadIdx.x;
    if (i < n) deg[i] = rsqrtf(deg[i]);
}

// -------- weight transpose + f32 -> bf16 convert: Wt[n*256+k] = W[k*256+n] --------
__global__ void wt_bf16_k(const float* __restrict__ W, __bf16* __restrict__ Wt) {
    int idx = blockIdx.x * blockDim.x + threadIdx.x;   // 0 .. 65535
    int k = idx >> 8;
    int n = idx & 255;
    Wt[n * D + k] = (__bf16)W[k * D + n];
}

// ---------------- WMMA GEMM: C[M,256] = A[M,256] * B[256,256] (+bias, relu) ----------------
// A: f32 row-major, converted to bf16 in-register. Bt: bf16, B transposed (Bt[n][k]).
// 1 wave -> one 16x16 tile of C. 16 waves/block cover all N. grid.x = M/16 = 1250.
__global__ __launch_bounds__(512) void gemm_wmma_k(
    const float* __restrict__ A, const __bf16* __restrict__ Bt,
    const float* __restrict__ bias, float* __restrict__ C, int relu) {

    const int lane  = threadIdx.x & 31;
    const int wave  = threadIdx.x >> 5;    // 0..15 -> N tile
    const int mtile = blockIdx.x;          // 0..1249
    const int hi    = lane >> 4;           // lane half
    const int lo    = lane & 15;

    const int arow  = mtile * 16 + lo;     // A row for this lane
    const int abase = hi * 16;             // A K-offset inside 32-chunk (contiguous, matches B)
    const int ncol  = wave * 16 + lo;      // C/B column for this lane
    const int bbase = hi * 16;             // B K-offset inside 32-chunk (contiguous, matches A)

    const float*  Ar = A  + (size_t)arow * D;
    const __bf16* Br = Bt + (size_t)ncol * D;

    v8f acc = {};
#pragma unroll
    for (int kk = 0; kk < D; kk += 32) {
        // A fragment: 16 contiguous f32 at kk+abase
        const float4* ap0 = (const float4*)(Ar + kk + abase);
        const float4* ap1 = (const float4*)(Ar + kk + abase + 8);
        float4 f0 = ap0[0], f1 = ap0[1];
        float4 f2 = ap1[0], f3 = ap1[1];
        v16bf a;
        a[0]  = (__bf16)f0.x; a[1]  = (__bf16)f0.y; a[2]  = (__bf16)f0.z; a[3]  = (__bf16)f0.w;
        a[4]  = (__bf16)f1.x; a[5]  = (__bf16)f1.y; a[6]  = (__bf16)f1.z; a[7]  = (__bf16)f1.w;
        a[8]  = (__bf16)f2.x; a[9]  = (__bf16)f2.y; a[10] = (__bf16)f2.z; a[11] = (__bf16)f2.w;
        a[12] = (__bf16)f3.x; a[13] = (__bf16)f3.y; a[14] = (__bf16)f3.z; a[15] = (__bf16)f3.w;

        // B fragment: 16 contiguous bf16 of column ncol starting at K = kk + bbase
        v16bf b = *(const v16bf*)(Br + kk + bbase);

        acc = __builtin_amdgcn_wmma_f32_16x16x32_bf16(
                  /*neg_a=*/false, a, /*neg_b=*/false, b,
                  /*c_mod=*/(short)0, acc, /*reuse_a=*/false, /*reuse_b=*/false);
    }

    const float bv = bias ? bias[ncol] : 0.0f;
#pragma unroll
    for (int v = 0; v < 8; ++v) {
        float val = acc[v] + bv;                      // C layout: M = v + 8*hi, N = ncol
        if (relu) val = fmaxf(val, 0.0f);
        C[(size_t)(mtile * 16 + v + 8 * hi) * D + ncol] = val;
    }
}

// ------------- scatter init: out[i][f] = b[f] + dinv[i]^2 * t[i][f]  (self loop) -------------
__global__ void scatter_init_k(const float* __restrict__ t, const float* __restrict__ dinv,
                               const float* __restrict__ bias, float* __restrict__ out) {
    int idx = blockIdx.x * blockDim.x + threadIdx.x;   // node*64 + f4
    if (idx >= NNODES * (D / 4)) return;
    int i = idx >> 6;
    int f = idx & 63;
    float d = dinv[i];
    float w = d * d;
    float4 tv = ((const float4*)t)[idx];
    float4 bv = ((const float4*)bias)[f];
    float4 o;
    o.x = bv.x + w * tv.x;
    o.y = bv.y + w * tv.y;
    o.z = bv.z + w * tv.z;
    o.w = bv.w + w * tv.w;
    ((float4*)out)[idx] = o;
}

// ------------- scatter edges: out[dst] += dinv[src]*dinv[dst]*t[src] (1 warp / edge) -------------
__global__ void scatter_edges_k(const int* __restrict__ src, const int* __restrict__ dst,
                                const float* __restrict__ dinv, const float* __restrict__ t,
                                float* __restrict__ out, int e) {
    int eid = blockIdx.x * 8 + (threadIdx.x >> 5);
    if (eid >= e) return;
    int lane = threadIdx.x & 31;
    int s  = src[eid];
    int d2 = dst[eid];
    float w = dinv[s] * dinv[d2];
    const float4* tp = (const float4*)(t + (size_t)s * D) + lane * 2;
    float4 x0 = tp[0];
    float4 x1 = tp[1];
    float* op = out + (size_t)d2 * D + lane * 8;
    atomicAdd(op + 0, w * x0.x);
    atomicAdd(op + 1, w * x0.y);
    atomicAdd(op + 2, w * x0.z);
    atomicAdd(op + 3, w * x0.w);
    atomicAdd(op + 4, w * x1.x);
    atomicAdd(op + 5, w * x1.y);
    atomicAdd(op + 6, w * x1.z);
    atomicAdd(op + 7, w * x1.w);
}

// ------------- classifier: logits[i][c] = h[i] . Wc[:,c] + bc[c], c < 3 -------------
__global__ void cls_k(const float* __restrict__ h, const float* __restrict__ Wc,
                      const float* __restrict__ bc, float* __restrict__ out) {
    int idx = blockIdx.x * blockDim.x + threadIdx.x;
    if (idx >= NNODES * 3) return;
    int i = idx / 3;
    int c = idx - i * 3;
    float acc = bc[c];
    const float* hr = h + (size_t)i * D;
#pragma unroll 8
    for (int k = 0; k < D; ++k) acc += hr[k] * Wc[k * 3 + c];
    out[idx] = acc;
}

extern "C" void kernel_launch(void* const* d_in, const int* in_sizes, int n_in,
                              void* d_out, int out_size, void* d_ws, size_t ws_size,
                              hipStream_t stream) {
    (void)in_sizes; (void)n_in; (void)out_size; (void)ws_size;

    const float* x    = (const float*)d_in[0];
    const int*   ei   = (const int*)d_in[1];        // [2, NEDGES] row-major (int32 per harness)
    const int*   srcA = ei;
    const int*   dstA = ei + NEDGES;
    const float* We   = (const float*)d_in[2];
    const float* be   = (const float*)d_in[3];
    const float* W1   = (const float*)d_in[4];
    const float* b1   = (const float*)d_in[5];
    const float* W2   = (const float*)d_in[6];
    const float* b2   = (const float*)d_in[7];
    const float* Wc   = (const float*)d_in[8];
    const float* bc   = (const float*)d_in[9];
    float* out = (float*)d_out;

    // workspace carve-up (~41.5 MB)
    char* ws = (char*)d_ws;
    size_t off = 0;
    float* dinv = (float*)(ws + off); off += (((size_t)NNODES * 4) + 255) & ~(size_t)255;
    float* bufA = (float*)(ws + off); off += (size_t)NNODES * D * 4;
    float* bufB = (float*)(ws + off); off += (size_t)NNODES * D * 4;
    __bf16* WtE = (__bf16*)(ws + off); off += (size_t)D * D * 2;
    __bf16* Wt1 = (__bf16*)(ws + off); off += (size_t)D * D * 2;
    __bf16* Wt2 = (__bf16*)(ws + off); off += (size_t)D * D * 2;

    // 1) symmetric normalization coefficients
    deg_init_k <<<(NNODES + 255) / 256, 256, 0, stream>>>(dinv, NNODES);
    deg_edges_k<<<(NEDGES + 255) / 256, 256, 0, stream>>>(dstA, dinv, NEDGES);
    rsqrt_k    <<<(NNODES + 255) / 256, 256, 0, stream>>>(dinv, NNODES);

    // 2) weights -> transposed bf16
    wt_bf16_k<<<(D * D) / 256, 256, 0, stream>>>(We, WtE);
    wt_bf16_k<<<(D * D) / 256, 256, 0, stream>>>(W1, Wt1);
    wt_bf16_k<<<(D * D) / 256, 256, 0, stream>>>(W2, Wt2);

    const int MT = NNODES / 16;   // 1250 M tiles (20000 % 16 == 0)

    // 3) h0 = relu(x @ We + be)
    gemm_wmma_k<<<MT, 512, 0, stream>>>(x, WtE, be, bufA, 1);

    // 4) layer 1: t1 = h0 @ W1 ; out1 = segment_sum(norm * t1[src]) + b1
    gemm_wmma_k   <<<MT, 512, 0, stream>>>(bufA, Wt1, nullptr, bufB, 0);
    scatter_init_k<<<(NNODES * (D / 4) + 255) / 256, 256, 0, stream>>>(bufB, dinv, b1, bufA);
    scatter_edges_k<<<(NEDGES + 7) / 8, 256, 0, stream>>>(srcA, dstA, dinv, bufB, bufA, NEDGES);

    // 5) layer 2
    gemm_wmma_k   <<<MT, 512, 0, stream>>>(bufA, Wt2, nullptr, bufB, 0);
    scatter_init_k<<<(NNODES * (D / 4) + 255) / 256, 256, 0, stream>>>(bufB, dinv, b2, bufA);
    scatter_edges_k<<<(NEDGES + 7) / 8, 256, 0, stream>>>(srcA, dstA, dinv, bufB, bufA, NEDGES);

    // 6) logits
    cls_k<<<(NNODES * 3 + 255) / 256, 256, 0, stream>>>(bufA, Wc, bc, out);
}
